// HGNNModel_23244363006023
// MI455X (gfx1250) — matrix operand order repaired and checked by reading, not compiled, not run on hardware
//
#include <hip/hip_runtime.h>
#include <hip/hip_bf16.h>

// ---------------- CDNA5 WMMA types ----------------
typedef __attribute__((ext_vector_type(16))) __bf16 v16bf;
typedef __attribute__((ext_vector_type(8)))  __bf16 v8bf;
typedef __attribute__((ext_vector_type(8)))  float  v8f;

#define N_NODES 32768
#define N_EDGES 32768
#define NNZ     262144
#define N_GRAPHS 128
#define MAX_NODES 256
#define IN_DIM  300
#define LD0     304        // padded f32 stride for 300-dim arrays
#define KP1     320        // layer-1 K padded to multiple of 32 (bf16 GEMM)
#define HID     256
#define N_CLASSES 20

// ---------------- utility kernels ----------------
__global__ void fill_zero_kernel(float* p, size_t n) {
  size_t stride = (size_t)gridDim.x * blockDim.x;
  for (size_t i = (size_t)blockIdx.x * blockDim.x + threadIdx.x; i < n; i += stride)
    p[i] = 0.0f;
}

__global__ void init_u32_kernel(unsigned* p, unsigned v) {
  if (threadIdx.x == 0 && blockIdx.x == 0) *p = v;
}

// h0[n, 0:300] = emb[x[n], :]; pad cols 300..303 with zero
__global__ void gather_emb_kernel(const int* __restrict__ x,
                                  const float* __restrict__ emb,
                                  float* __restrict__ h0) {
  size_t total = (size_t)N_NODES * LD0;
  size_t stride = (size_t)gridDim.x * blockDim.x;
  for (size_t t = (size_t)blockIdx.x * blockDim.x + threadIdx.x; t < total; t += stride) {
    int n = (int)(t / LD0), d = (int)(t % LD0);
    h0[t] = (d < IN_DIM) ? emb[(size_t)x[n] * IN_DIM + d] : 0.0f;
  }
}

__global__ void count_deg_kernel(const int* __restrict__ rows,
                                 const int* __restrict__ cols,
                                 float* __restrict__ dv, float* __restrict__ de) {
  size_t stride = (size_t)gridDim.x * blockDim.x;
  for (size_t i = (size_t)blockIdx.x * blockDim.x + threadIdx.x; i < NNZ; i += stride) {
    atomicAdd(&dv[rows[i]], 1.0f);
    atomicAdd(&de[cols[i]], 1.0f);
  }
}

__global__ void inv_deg_kernel(const float* __restrict__ d, float* __restrict__ dinv, int n) {
  int i = blockIdx.x * blockDim.x + threadIdx.x;
  if (i < n) { float v = d[i]; dinv[i] = (v > 0.0f) ? 1.0f / v : 0.0f; }
}

// hout[dst[i]] += vals[i] * hin[src[i]]  (per feature dim)
__global__ void spmm_scatter_kernel(const int* __restrict__ src_idx,
                                    const int* __restrict__ dst_idx,
                                    const float* __restrict__ vals,
                                    const float* __restrict__ hin,
                                    float* __restrict__ hout,
                                    int D, int ld) {
  int chunks = (D + 3) >> 2;
  size_t total = (size_t)NNZ * chunks;
  size_t stride = (size_t)gridDim.x * blockDim.x;
  for (size_t t = (size_t)blockIdx.x * blockDim.x + threadIdx.x; t < total; t += stride) {
    int i = (int)(t / chunks);
    int d0 = ((int)(t % chunks)) << 2;
    float v = vals[i];
    const float* src = hin + (size_t)src_idx[i] * ld + d0;
    float* dst = hout + (size_t)dst_idx[i] * ld + d0;
    int rem = D - d0; rem = rem > 4 ? 4 : rem;
    for (int q = 0; q < rem; ++q) atomicAdd(dst + q, v * src[q]);
  }
}

__global__ void scale_rows_kernel(float* __restrict__ h, const float* __restrict__ inv,
                                  int rows, int D, int ld) {
  size_t total = (size_t)rows * D;
  size_t stride = (size_t)gridDim.x * blockDim.x;
  for (size_t t = (size_t)blockIdx.x * blockDim.x + threadIdx.x; t < total; t += stride) {
    int r = (int)(t / D), d = (int)(t % D);
    h[(size_t)r * ld + d] *= inv[r];
  }
}

// W (f32, [K x HID] row-major) -> bf16 [Kp x HID], zero-padded rows K..Kp-1
__global__ void convert_w_bf16_kernel(const float* __restrict__ W, __bf16* __restrict__ Wb,
                                      int K, int Kp) {
  size_t total = (size_t)Kp * HID;
  size_t stride = (size_t)gridDim.x * blockDim.x;
  for (size_t t = (size_t)blockIdx.x * blockDim.x + threadIdx.x; t < total; t += stride) {
    int r = (int)(t / HID), c = (int)(t % HID);
    Wb[t] = (r < K) ? (__bf16)W[(size_t)r * HID + c] : (__bf16)0.0f;
  }
}

// A = bf16(h * inv[row]) with K padded: h f32 [rows x D] stride ld -> bf16 [rows x Kp]
__global__ void scale_convert_bf16_kernel(const float* __restrict__ h, int ld, int D,
                                          const float* __restrict__ inv,
                                          __bf16* __restrict__ out, int Kp, int rows) {
  size_t total = (size_t)rows * Kp;
  size_t stride = (size_t)gridDim.x * blockDim.x;
  for (size_t t = (size_t)blockIdx.x * blockDim.x + threadIdx.x; t < total; t += stride) {
    int r = (int)(t / Kp), d = (int)(t % Kp);
    out[t] = (d < D) ? (__bf16)(h[(size_t)r * ld + d] * inv[r]) : (__bf16)0.0f;
  }
}

// ---------------- WMMA GEMM: C = relu(A[M x Kp]bf16 @ B[Kp x 256]bf16 + bias) ----------------
// Each wave computes a 16(M) x 64(N) strip: one shared A fragment, 4 accumulators.
// All fragment loads for a k-step are issued as one batch (2x b128 for A, 64 strided 16-bit loads for B)
// before the 4-WMMA chain so load latency overlaps with matrix-pipe execution.
__global__ void gemm_bias_relu_wmma(const __bf16* __restrict__ A, int Kp,
                                    const __bf16* __restrict__ B,
                                    const float* __restrict__ bias,
                                    float* __restrict__ C, int M) {
  const int NGRP = HID / 64; // 4 n-groups of 64
  int wave = blockIdx.x * (blockDim.x >> 5) + (threadIdx.x >> 5);
  int tm = wave / NGRP, ng = wave % NGRP;
  if (tm >= (M >> 4)) return;
  int lane = threadIdx.x & 31;
  int l16 = lane & 15;     // A row in tile / C column in subtile
  int half = lane >> 4;
  int m0 = tm << 4, n0 = ng << 6;

  const v8bf* Ap = (const v8bf*)(A + (size_t)(m0 + l16) * Kp);
  v8f acc[4];
#pragma unroll
  for (int j = 0; j < 4; ++j) acc[j] = (v8f){0.f,0.f,0.f,0.f,0.f,0.f,0.f,0.f};

  for (int k0 = 0; k0 < Kp; k0 += 32) {
    // ---- batch all loads for this k-step ----
    v8bf alo = Ap[(k0 >> 3) + half];
    v8bf ahi = Ap[(k0 >> 3) + 2 + half];
    const __bf16* Bp = B + (size_t)k0 * HID + n0 + l16;
    v8bf blo[4], bhi[4];
#pragma unroll
    for (int j = 0; j < 4; ++j) {
#pragma unroll
      for (int s = 0; s < 8; ++s) {
        blo[j][s] = Bp[(size_t)((half << 3) + s) * HID + (j << 4)];
        bhi[j][s] = Bp[(size_t)(16 + (half << 3) + s) * HID + (j << 4)];
      }
    }
    // ---- WMMA chain ----
    v16bf af = __builtin_shufflevector(alo, ahi,
        0,1,2,3,4,5,6,7,8,9,10,11,12,13,14,15);
#pragma unroll
    for (int j = 0; j < 4; ++j) {
      v16bf bf = __builtin_shufflevector(blo[j], bhi[j],
          0,1,2,3,4,5,6,7,8,9,10,11,12,13,14,15);
      acc[j] = __builtin_amdgcn_wmma_f32_16x16x32_bf16(
          /*neg_a=*/false, af, /*neg_b=*/false, bf,
          /*c_mod=*/(short)0, acc[j], /*reuse_a=*/false, /*reuse_b=*/false);
    }
  }
  // C layout: n = lane%16, m = vgpr + 8*(lane/16)
#pragma unroll
  for (int j = 0; j < 4; ++j) {
    int n = n0 + (j << 4) + l16;
    float bv = bias[n];
#pragma unroll
    for (int r = 0; r < 8; ++r) {
      int m = (half << 3) + r;
      float v = acc[j][r] + bv;
      C[(size_t)(m0 + m) * HID + n] = (v > 0.0f) ? v : 0.0f;
    }
  }
}

// ---------------- pooling ----------------
// s[n] = act(h[n,:].aw + ab); one wave32 per node
__global__ void score_kernel(const float* __restrict__ h, int ld, int D,
                             const float* __restrict__ aw, const float* __restrict__ ab,
                             int use_tanh, float* __restrict__ s) {
  int wid = (blockIdx.x * blockDim.x + threadIdx.x) >> 5;
  int lane = threadIdx.x & 31;
  if (wid >= N_NODES) return;
  const float* hp = h + (size_t)wid * ld;
  float acc = 0.0f;
  for (int d = lane; d < D; d += 32) acc += hp[d] * aw[d];
  for (int off = 16; off > 0; off >>= 1) acc += __shfl_down(acc, off);
  if (lane == 0) {
    float v = acc + ab[0];
    if (use_tanh) v = tanhf(v);
    s[wid] = v;
  }
}

__device__ inline unsigned float_map_u32(float f) {
  unsigned b = __float_as_uint(f);
  return (b & 0x80000000u) ? ~b : (b | 0x80000000u);
}
__device__ inline float float_unmap_u32(unsigned u) {
  unsigned b = (u & 0x80000000u) ? (u ^ 0x80000000u) : ~u;
  return __uint_as_float(b);
}

__global__ void max_reduce_kernel(const float* __restrict__ s, int n, unsigned* __restrict__ out) {
  __shared__ float red[256];
  float m = -3.4e38f;
  size_t stride = (size_t)gridDim.x * blockDim.x;
  for (size_t i = (size_t)blockIdx.x * blockDim.x + threadIdx.x; i < (size_t)n; i += stride)
    m = fmaxf(m, s[i]);
  red[threadIdx.x] = m;
  __syncthreads();
  for (int off = 128; off > 0; off >>= 1) {
    if ((int)threadIdx.x < off) red[threadIdx.x] = fmaxf(red[threadIdx.x], red[threadIdx.x + off]);
    __syncthreads();
  }
  if (threadIdx.x == 0) atomicMax(out, float_map_u32(red[0]));
}

// one 256-thread block per graph; graphs are contiguous 256-node spans
__global__ void pool_kernel(const float* __restrict__ h, int ld, int D,
                            const float* __restrict__ s, const unsigned* __restrict__ maxbuf,
                            float* __restrict__ g) {
  __shared__ float es[MAX_NODES];
  __shared__ float red[MAX_NODES];
  __shared__ float ssum;
  int gi = blockIdx.x, t = threadIdx.x;
  float M = float_unmap_u32(*maxbuf);
  float e = expf(s[(size_t)gi * MAX_NODES + t] - M);
  es[t] = e; red[t] = e;
  __syncthreads();
  for (int off = 128; off > 0; off >>= 1) {
    if (t < off) red[t] += red[t + off];
    __syncthreads();
  }
  if (t == 0) ssum = red[0];
  __syncthreads();
  float inv = 1.0f / (ssum + 1e-10f);
  const float* hb = h + (size_t)gi * MAX_NODES * ld;
  for (int d = t; d < D; d += 256) {
    float acc = 0.0f, mx = -1.0e9f;
    for (int m = 0; m < MAX_NODES; ++m) {
      float v = hb[(size_t)m * ld + d];
      acc += es[m] * v;
      mx = fmaxf(mx, v);
    }
    g[(size_t)gi * D + d] = acc * inv + mx;
  }
}

// ---------------- final projection + outputs ----------------
__global__ void final_kernel(const float* __restrict__ g0, const float* __restrict__ g1,
                             const float* __restrict__ g2,
                             const float* __restrict__ p0w, const float* __restrict__ p0b,
                             const float* __restrict__ p1w, const float* __restrict__ p1b,
                             const float* __restrict__ p2w, const float* __restrict__ p2b,
                             const int* __restrict__ targets, float* __restrict__ out) {
  int g = blockIdx.x, t = threadIdx.x; // blockDim = 64
  if (t < N_CLASSES) {
    float acc = p0b[t] + p1b[t] + p2b[t];
    const float* g0p = g0 + (size_t)g * IN_DIM;
    const float* g1p = g1 + (size_t)g * HID;
    const float* g2p = g2 + (size_t)g * HID;
    for (int d = 0; d < IN_DIM; ++d) acc += g0p[d] * p0w[d * N_CLASSES + t];
    for (int d = 0; d < HID;   ++d) acc += g1p[d] * p1w[d * N_CLASSES + t];
    for (int d = 0; d < HID;   ++d) acc += g2p[d] * p2w[d * N_CLASSES + t];
    out[g * N_CLASSES + t] = acc;
  }
  if (t == 32) out[N_GRAPHS * N_CLASSES + g] = (float)targets[g];
  float* o2 = out + N_GRAPHS * N_CLASSES + N_GRAPHS;
  for (int d = t; d < HID; d += 64) o2[(size_t)g * HID + d] = g2[(size_t)g * HID + d];
}

// ---------------- host launcher ----------------
extern "C" void kernel_launch(void* const* d_in, const int* in_sizes, int n_in,
                              void* d_out, int out_size, void* d_ws, size_t ws_size,
                              hipStream_t stream) {
  (void)in_sizes; (void)n_in; (void)out_size; (void)ws_size;
  const int*   x    = (const int*)d_in[0];
  const int*   rows = (const int*)d_in[1];
  const int*   cols = (const int*)d_in[2];
  const float* vals = (const float*)d_in[3];
  // d_in[4] max_pool_idx: arange -> graphs contiguous, unused
  const int*   targets = (const int*)d_in[5];
  const float* emb = (const float*)d_in[6];
  const float* W1  = (const float*)d_in[7];
  const float* b1  = (const float*)d_in[8];
  const float* W2  = (const float*)d_in[9];
  const float* b2  = (const float*)d_in[10];
  const float* a0w = (const float*)d_in[11]; const float* a0b = (const float*)d_in[12];
  const float* a1w = (const float*)d_in[13]; const float* a1b = (const float*)d_in[14];
  const float* a2w = (const float*)d_in[15]; const float* a2b = (const float*)d_in[16];
  const float* p0w = (const float*)d_in[17]; const float* p0b = (const float*)d_in[18];
  const float* p1w = (const float*)d_in[19]; const float* p1b = (const float*)d_in[20];
  const float* p2w = (const float*)d_in[21]; const float* p2b = (const float*)d_in[22];
  float* out = (float*)d_out;
  float* ws  = (float*)d_ws;

  // workspace layout (float offsets; bf16 buffers counted as half-floats, 16B aligned)
  const size_t off_h0  = 0;                               // N*LD0 (reused for h2, ld=HID)
  const size_t off_he  = off_h0 + (size_t)N_NODES * LD0;  // E*LD0 (reused for he2)
  const size_t off_hv  = off_he + (size_t)N_EDGES * LD0;  // N*LD0 (reused for hv2)
  const size_t off_h1  = off_hv + (size_t)N_NODES * LD0;  // N*HID
  const size_t off_dv  = off_h1 + (size_t)N_NODES * HID;
  const size_t off_de  = off_dv + N_NODES;
  const size_t off_dvi = off_de + N_EDGES;
  const size_t off_dei = off_dvi + N_NODES;
  const size_t off_s   = off_dei + N_EDGES;
  const size_t off_max = off_s + N_NODES;                 // 16 floats reserved
  const size_t off_g0  = off_max + 16;                    // G*300
  const size_t off_g1  = off_g0 + (size_t)N_GRAPHS * IN_DIM;
  const size_t off_g2  = off_g1 + (size_t)N_GRAPHS * HID;
  size_t off_abf = off_g2 + (size_t)N_GRAPHS * HID;
  off_abf = (off_abf + 15) & ~(size_t)15;                 // bf16 A: N*KP1 elems
  size_t off_wb  = off_abf + ((size_t)N_NODES * KP1) / 2;
  off_wb = (off_wb + 15) & ~(size_t)15;                   // bf16 W: KP1*HID elems

  float* h0 = ws + off_h0;
  float* he = ws + off_he;
  float* hv = ws + off_hv;
  float* h1 = ws + off_h1;
  float* h2 = ws + off_h0;   // reuse h0 region after g0/layer1 consumed
  float* sb = ws + off_s;
  unsigned* maxbuf = (unsigned*)(ws + off_max);
  __bf16* Abf = (__bf16*)(ws + off_abf);
  __bf16* Wb  = (__bf16*)(ws + off_wb);
  const unsigned MAP_NEG_INF = 0x007FFFFFu; // mapped(-inf)

  const int B256 = 256;
  const int GB_FILL = 4096, GB_SPMM = 8192;
  const int GEMM_BLOCKS = (N_NODES / 16) * (HID / 64) / 8; // 8 waves/block, 16x64 per wave

  // degrees
  fill_zero_kernel<<<256, B256, 0, stream>>>(ws + off_dv, (size_t)(N_NODES + N_EDGES));
  count_deg_kernel<<<1024, B256, 0, stream>>>(rows, cols, ws + off_dv, ws + off_de);
  inv_deg_kernel<<<N_NODES / B256, B256, 0, stream>>>(ws + off_dv, ws + off_dvi, N_NODES);
  inv_deg_kernel<<<N_EDGES / B256, B256, 0, stream>>>(ws + off_de, ws + off_dei, N_EDGES);

  // h0 = emb[x]
  gather_emb_kernel<<<GB_FILL, B256, 0, stream>>>(x, emb, h0);

  // pool g0 (identity act)
  score_kernel<<<(N_NODES * 32) / B256, B256, 0, stream>>>(h0, LD0, IN_DIM, a0w, a0b, 0, sb);
  init_u32_kernel<<<1, 32, 0, stream>>>(maxbuf, MAP_NEG_INF);
  max_reduce_kernel<<<256, B256, 0, stream>>>(sb, N_NODES, maxbuf);
  pool_kernel<<<N_GRAPHS, B256, 0, stream>>>(h0, LD0, IN_DIM, sb, maxbuf, ws + off_g0);

  // layer 1: he = De^-1 H^T h0 ; hv = Dv^-1 H he ; h1 = relu(hv @ W1 + b1)
  fill_zero_kernel<<<GB_FILL, B256, 0, stream>>>(he, (size_t)N_EDGES * LD0);
  spmm_scatter_kernel<<<GB_SPMM, B256, 0, stream>>>(rows, cols, vals, h0, he, IN_DIM, LD0);
  scale_rows_kernel<<<GB_FILL, B256, 0, stream>>>(he, ws + off_dei, N_EDGES, IN_DIM, LD0);
  fill_zero_kernel<<<GB_FILL, B256, 0, stream>>>(hv, (size_t)N_NODES * LD0);
  spmm_scatter_kernel<<<GB_SPMM, B256, 0, stream>>>(cols, rows, vals, he, hv, IN_DIM, LD0);
  // fold dv_inv scaling into bf16 conversion of the GEMM A operand
  scale_convert_bf16_kernel<<<GB_FILL, B256, 0, stream>>>(hv, LD0, IN_DIM, ws + off_dvi,
                                                          Abf, KP1, N_NODES);
  convert_w_bf16_kernel<<<320, B256, 0, stream>>>(W1, Wb, IN_DIM, KP1);
  gemm_bias_relu_wmma<<<GEMM_BLOCKS, B256, 0, stream>>>(Abf, KP1, Wb, b1, h1, N_NODES);

  // pool g1 (tanh act)
  score_kernel<<<(N_NODES * 32) / B256, B256, 0, stream>>>(h1, HID, HID, a1w, a1b, 1, sb);
  init_u32_kernel<<<1, 32, 0, stream>>>(maxbuf, MAP_NEG_INF);
  max_reduce_kernel<<<256, B256, 0, stream>>>(sb, N_NODES, maxbuf);
  pool_kernel<<<N_GRAPHS, B256, 0, stream>>>(h1, HID, HID, sb, maxbuf, ws + off_g1);

  // layer 2 (reuse he/hv regions with ld = HID)
  fill_zero_kernel<<<GB_FILL, B256, 0, stream>>>(he, (size_t)N_EDGES * HID);
  spmm_scatter_kernel<<<GB_SPMM, B256, 0, stream>>>(rows, cols, vals, h1, he, HID, HID);
  scale_rows_kernel<<<GB_FILL, B256, 0, stream>>>(he, ws + off_dei, N_EDGES, HID, HID);
  fill_zero_kernel<<<GB_FILL, B256, 0, stream>>>(hv, (size_t)N_NODES * HID);
  spmm_scatter_kernel<<<GB_SPMM, B256, 0, stream>>>(cols, rows, vals, he, hv, HID, HID);
  scale_convert_bf16_kernel<<<GB_FILL, B256, 0, stream>>>(hv, HID, HID, ws + off_dvi,
                                                          Abf, HID, N_NODES);
  convert_w_bf16_kernel<<<256, B256, 0, stream>>>(W2, Wb, HID, HID);
  gemm_bias_relu_wmma<<<GEMM_BLOCKS, B256, 0, stream>>>(Abf, HID, Wb, b2, h2, N_NODES);

  // pool g2 (identity act)
  score_kernel<<<(N_NODES * 32) / B256, B256, 0, stream>>>(h2, HID, HID, a2w, a2b, 0, sb);
  init_u32_kernel<<<1, 32, 0, stream>>>(maxbuf, MAP_NEG_INF);
  max_reduce_kernel<<<256, B256, 0, stream>>>(sb, N_NODES, maxbuf);
  pool_kernel<<<N_GRAPHS, B256, 0, stream>>>(h2, HID, HID, sb, maxbuf, ws + off_g2);

  // pred + targets + g2 -> d_out
  final_kernel<<<N_GRAPHS, 64, 0, stream>>>(ws + off_g0, ws + off_g1, ws + off_g2,
                                            p0w, p0b, p1w, p1b, p2w, p2b, targets, out);
}
